// Block_74354473828562
// MI455X (gfx1250) — compile-verified, matrix operand order checked by reading, not run on hardware
//
#include <hip/hip_runtime.h>

// ---------------- types ----------------
typedef _Float16 h8   __attribute__((ext_vector_type(8)));
typedef _Float16 v16h __attribute__((ext_vector_type(16)));
typedef float    v8f  __attribute__((ext_vector_type(8)));

// problem constants
#define BB 4
#define SS 512
#define DD 768
#define HH 12
#define MT 4
#define DFF 3072
#define HD 64
#define RR (BB*SS)      // 2048 flat rows
#define BH (BB*HH)      // 48
#define SCALE_ATT 0.125f

// gemm tiling: all launched shapes satisfy M%128==0, N%64==0, K%64==0 (no bounds checks)
#define BM 128
#define BN 64
#define BK 64

// epilogue flags
#define EPI_BIAS   1
#define EPI_RESID  2
#define EPI_GELU   4
#define EPI_GPRIME 8
#define EPI_SCALE  16
#define EPI_ACC    32

__device__ __forceinline__ float gelu_f(float x) {
    return 0.5f * x * (1.0f + erff(x * 0.70710678118654752f));
}
__device__ __forceinline__ float gelu_grad(float x) {
    float cdf = 0.5f * (1.0f + erff(x * 0.70710678118654752f));
    float pdf = 0.3989422804014327f * expf(-0.5f * x * x);
    return cdf + x * pdf;
}

struct GemmParams {
    const _Float16* A;      // [M,K] row-major (+ z offset)
    const _Float16* B;      // Bt: [N,K] row-major (+ z offset)
    const float* bias;      // [N]
    const float* resid;     // f32 residual, ldr
    const float* U;         // pre-activation for GELU' (flat only)
    float* Cf;              // f32 out / accumulate source
    _Float16* Ch;           // f16 out
    int M, N, K;
    int lda, ldb, ldcf, ldch, ldr, ldu;
    int zdiv;
    long long sA0, sA1, sB0, sB1, sCf0, sCf1, sCh0, sCh1, sR0, sR1;
    float scale;
};

// async global->LDS copy of one 16-byte chunk (CDNA5 ASYNCcnt path)
__device__ __forceinline__ void async_copy16(const _Float16* gsrc, _Float16* ldst) {
    unsigned lds_off = (unsigned)(unsigned long long)ldst; // LDS aperture: addr[31:0] == LDS offset
    asm volatile("global_load_async_to_lds_b128 %0, %1, off"
                 :: "v"(lds_off), "v"((unsigned long long)gsrc)
                 : "memory");
}

template <int EPI, bool OUTF, bool OUTH>
__global__ __launch_bounds__(256) void gemm_kernel(GemmParams p) {
    __shared__ __align__(16) _Float16 sA[2][BM][BK];  // 32 KB
    __shared__ __align__(16) _Float16 sB[2][BN][BK];  // 16 KB

    const int tid  = threadIdx.x;
    const int z    = blockIdx.z;
    const long long zq = z / p.zdiv;
    const long long zr = z % p.zdiv;
    const _Float16* Ag = p.A + zq * p.sA0 + zr * p.sA1;
    const _Float16* Bg = p.B + zq * p.sB0 + zr * p.sB1;
    const long long cfOff = zq * p.sCf0 + zr * p.sCf1;
    const long long chOff = zq * p.sCh0 + zr * p.sCh1;
    const long long rOff  = zq * p.sR0  + zr * p.sR1;

    const int m0 = blockIdx.y * BM;
    const int n0 = blockIdx.x * BN;
    const int wave = tid >> 5, lane = tid & 31;
    const int wm = wave >> 1, wn = wave & 1;
    const int lm = lane & 15, hi = lane >> 4;
    const int kb = hi * 8;

    // per-thread staging chunks: A = 128*64/8 = 1024 chunks (4/thread), B = 64*64/8 = 512 (2/thread)
    const int aRow[4] = { (0*256 + tid) >> 3, (1*256 + tid) >> 3, (2*256 + tid) >> 3, (3*256 + tid) >> 3 };
    const int aKc     = (tid & 7) * 8;
    const int bRow[2] = { (0*256 + tid) >> 3, (1*256 + tid) >> 3 };

    auto issue = [&](int t, int buf) {
#pragma unroll
        for (int i = 0; i < 4; ++i)
            async_copy16(Ag + (long long)(m0 + aRow[i]) * p.lda + t * BK + aKc,
                         &sA[buf][aRow[i]][aKc]);
#pragma unroll
        for (int i = 0; i < 2; ++i)
            async_copy16(Bg + (long long)(n0 + bRow[i]) * p.ldb + t * BK + aKc,
                         &sB[buf][bRow[i]][aKc]);
    };

    v8f acc[2][2] = {};
    const int T = p.K >> 6;   // K / 64, exact

    issue(0, 0);
    for (int t = 0; t < T; ++t) {
        const int cur = t & 1;
        if (t + 1 < T) {
            issue(t + 1, cur ^ 1);
            // 6 asyncs/wave/tile: <=6 outstanding means tile t landed, t+1 still in flight
            asm volatile("s_wait_asynccnt 0x6" ::: "memory");
        } else {
            asm volatile("s_wait_asynccnt 0x0" ::: "memory");
        }
        __syncthreads();

#pragma unroll
        for (int ks = 0; ks < 2; ++ks) {
            const int kbase = ks * 32 + kb;
            v16h af[2], bf[2];
#pragma unroll
            for (int tt = 0; tt < 2; ++tt) {
                int r = wm * 32 + tt * 16 + lm;
                h8 lo  = *(const h8*)&sA[cur][r][kbase];
                h8 hi8 = *(const h8*)&sA[cur][r][kbase + 16];
                af[tt] = __builtin_shufflevector(lo, hi8, 0,1,2,3,4,5,6,7,8,9,10,11,12,13,14,15);
                int rn = wn * 32 + tt * 16 + lm;
                h8 blo = *(const h8*)&sB[cur][rn][kbase];
                h8 bhi = *(const h8*)&sB[cur][rn][kbase + 16];
                bf[tt] = __builtin_shufflevector(blo, bhi, 0,1,2,3,4,5,6,7,8,9,10,11,12,13,14,15);
            }
#pragma unroll
            for (int tm = 0; tm < 2; ++tm)
#pragma unroll
                for (int tn = 0; tn < 2; ++tn)
                    acc[tm][tn] = __builtin_amdgcn_wmma_f32_16x16x32_f16(
                        false, af[tm], false, bf[tn], (short)0, acc[tm][tn], false, false);
        }
        __syncthreads();
    }

    // epilogue (shapes tile exactly; no bounds checks)
#pragma unroll
    for (int tm = 0; tm < 2; ++tm) {
#pragma unroll
        for (int tn = 0; tn < 2; ++tn) {
#pragma unroll
            for (int r = 0; r < 8; ++r) {
                int gm = m0 + wm * 32 + tm * 16 + hi * 8 + r;
                int gn = n0 + wn * 32 + tn * 16 + lm;
                float v = acc[tm][tn][r];
                if (EPI & EPI_SCALE) v *= p.scale;
                if (EPI & EPI_BIAS)  v += p.bias[gn];
                if (EPI & EPI_ACC)   v += p.Cf[cfOff + (long long)gm * p.ldcf + gn];
                if (EPI & EPI_RESID) v += p.resid[rOff + (long long)gm * p.ldr + gn];
                float outv = v;
                if (EPI & EPI_GELU) {
                    if (OUTF) p.Cf[cfOff + (long long)gm * p.ldcf + gn] = v; // pre-act
                    outv = gelu_f(v);
                } else if (EPI & EPI_GPRIME) {
                    outv = v * gelu_grad(p.U[(long long)gm * p.ldu + gn]);
                }
                if (OUTF && !(EPI & EPI_GELU))
                    p.Cf[cfOff + (long long)gm * p.ldcf + gn] = outv;
                if (OUTH)
                    p.Ch[chOff + (long long)gm * p.ldch + gn] = (_Float16)outv;
            }
        }
    }
}

// ---------------- reductions / LN / softmax ----------------
__device__ __forceinline__ float block_reduce_sum(float v, float* red) {
    red[threadIdx.x] = v; __syncthreads();
    for (int o = blockDim.x >> 1; o > 0; o >>= 1) {
        if ((int)threadIdx.x < o) red[threadIdx.x] += red[threadIdx.x + o];
        __syncthreads();
    }
    float r = red[0]; __syncthreads();
    return r;
}
__device__ __forceinline__ float block_reduce_max(float v, float* red) {
    red[threadIdx.x] = v; __syncthreads();
    for (int o = blockDim.x >> 1; o > 0; o >>= 1) {
        if ((int)threadIdx.x < o) red[threadIdx.x] = fmaxf(red[threadIdx.x], red[threadIdx.x + o]);
        __syncthreads();
    }
    float r = red[0]; __syncthreads();
    return r;
}

__global__ __launch_bounds__(256) void ln_fwd_kernel(const float* X, const float* g, const float* b,
                                                     _Float16* Hh, float* Mu, float* Rs, int Dn) {
    __shared__ float red[256];
    long long base = (long long)blockIdx.x * Dn;
    float s = 0.f, s2 = 0.f;
    for (int i = threadIdx.x; i < Dn; i += 256) { float v = X[base + i]; s += v; s2 += v * v; }
    float mu  = block_reduce_sum(s, red) / Dn;
    float var = block_reduce_sum(s2, red) / Dn - mu * mu;
    float rs  = rsqrtf(var + 1e-6f);
    if (threadIdx.x == 0) { Mu[blockIdx.x] = mu; Rs[blockIdx.x] = rs; }
    for (int i = threadIdx.x; i < Dn; i += 256)
        Hh[base + i] = (_Float16)((X[base + i] - mu) * rs * g[i] + b[i]);
}

__global__ __launch_bounds__(256) void ln_jvp_kernel(const float* X, const float* dX,
                                                     const float* Mu, const float* Rs,
                                                     const float* g, _Float16* dH, int Dn) {
    __shared__ float red[256];
    long long base = (long long)blockIdx.x * Dn;
    float mu = Mu[blockIdx.x], rs = Rs[blockIdx.x];
    float s1 = 0.f, s2 = 0.f;
    for (int i = threadIdx.x; i < Dn; i += 256) {
        float dx = dX[base + i];
        float xh = (X[base + i] - mu) * rs;
        s1 += dx; s2 += dx * xh;
    }
    float m1 = block_reduce_sum(s1, red) / Dn;
    float m2 = block_reduce_sum(s2, red) / Dn;
    for (int i = threadIdx.x; i < Dn; i += 256) {
        float dx = dX[base + i];
        float xh = (X[base + i] - mu) * rs;
        dH[base + i] = (_Float16)(g[i] * rs * (dx - m1 - xh * m2));
    }
}

__global__ __launch_bounds__(128) void softmax_kernel(const float* S32, _Float16* P, int SN) {
    __shared__ float red[128];
    long long base = (long long)blockIdx.x * SN;
    float mx = -1e30f;
    for (int i = threadIdx.x; i < SN; i += 128) mx = fmaxf(mx, S32[base + i]);
    mx = block_reduce_max(mx, red);
    float sum = 0.f;
    for (int i = threadIdx.x; i < SN; i += 128) sum += expf(S32[base + i] - mx);
    sum = block_reduce_sum(sum, red);
    float inv = 1.0f / sum;
    for (int i = threadIdx.x; i < SN; i += 128)
        P[base + i] = (_Float16)(expf(S32[base + i] - mx) * inv);
}

__global__ __launch_bounds__(128) void softmax_jvp_kernel(const _Float16* P, const float* dS,
                                                          _Float16* dP, int SN) {
    __shared__ float red[128];
    long long base = (long long)blockIdx.x * SN;
    float s = 0.f;
    for (int i = threadIdx.x; i < SN; i += 128) s += (float)P[base + i] * dS[base + i];
    s = block_reduce_sum(s, red);
    for (int i = threadIdx.x; i < SN; i += 128)
        dP[base + i] = (_Float16)((float)P[base + i] * (dS[base + i] - s));
}

// transpose v slice: out[((b*H+h)*HD+d)*S + s] = QKV[(b*S+s)*ld + colBase + h*HD + d]
__global__ __launch_bounds__(256) void vtrans_kernel(const _Float16* QKV, _Float16* VT,
                                                     int ld, int colBase, long long total) {
    long long idx = (long long)blockIdx.x * 256 + threadIdx.x;
    if (idx >= total) return;
    int s = (int)(idx % SS);
    int d = (int)((idx / SS) % HD);
    int h = (int)((idx / ((long long)SS * HD)) % HH);
    int b = (int)(idx / ((long long)SS * HD * HH));
    VT[idx] = QKV[((long long)b * SS + s) * ld + colBase + h * HD + d];
}

// f32 W[K,N] -> f16 WT[N,K]
__global__ __launch_bounds__(256) void cvtT_kernel(const float* W, _Float16* WT,
                                                   int Kdim, int Ndim, long long total) {
    long long idx = (long long)blockIdx.x * 256 + threadIdx.x;
    if (idx >= total) return;
    int k = (int)(idx % Kdim);
    int n = (int)(idx / Kdim);
    WT[idx] = (_Float16)W[(long long)k * Ndim + n];
}

// gather tangent slice m: dst[(b*S+s)*D+d] = XT[((b*MT+m)*S+s)*D+d]
__global__ __launch_bounds__(256) void gather_kernel(const float* XT, float* dst, int m, long long total) {
    long long idx = (long long)blockIdx.x * 256 + threadIdx.x;
    if (idx >= total) return;
    int d = (int)(idx % DD);
    int s = (int)((idx / DD) % SS);
    int b = (int)(idx / ((long long)DD * SS));
    dst[idx] = XT[(((long long)b * MT + m) * SS + s) * DD + d];
}
__global__ __launch_bounds__(256) void scatter_kernel(const float* src, float* OT, int m, long long total) {
    long long idx = (long long)blockIdx.x * 256 + threadIdx.x;
    if (idx >= total) return;
    int d = (int)(idx % DD);
    int s = (int)((idx / DD) % SS);
    int b = (int)(idx / ((long long)DD * SS));
    OT[(((long long)b * MT + m) * SS + s) * DD + d] = src[idx];
}

// ---------------- host ----------------
extern "C" void kernel_launch(void* const* d_in, const int* in_sizes, int n_in,
                              void* d_out, int out_size, void* d_ws, size_t ws_size,
                              hipStream_t stream) {
    const float* x    = (const float*)d_in[0];
    const float* xt   = (const float*)d_in[1];
    const float* g1   = (const float*)d_in[2];
    const float* b1   = (const float*)d_in[3];
    const float* Wqkv = (const float*)d_in[4];
    const float* Wprj = (const float*)d_in[5];
    const float* bprj = (const float*)d_in[6];
    const float* g2   = (const float*)d_in[7];
    const float* b2   = (const float*)d_in[8];
    const float* W1   = (const float*)d_in[9];
    const float* bf1  = (const float*)d_in[10];
    const float* W2   = (const float*)d_in[11];
    const float* bf2  = (const float*)d_in[12];
    float* out = (float*)d_out;

    // workspace carve-out (~210 MB total)
    size_t off = 0;
    auto alloc = [&](size_t bytes) -> void* {
        void* p = (char*)d_ws + off;
        off += (bytes + 255) & ~(size_t)255;
        return p;
    };
    _Float16* wqkvT = (_Float16*)alloc((size_t)3 * DD * DD * 2);      // [2304][768]
    _Float16* wprjT = (_Float16*)alloc((size_t)DD * DD * 2);          // [768][768]
    _Float16* w1T   = (_Float16*)alloc((size_t)DFF * DD * 2);         // [3072][768]
    _Float16* w2T   = (_Float16*)alloc((size_t)DD * DFF * 2);         // [768][3072]
    _Float16* h16   = (_Float16*)alloc((size_t)RR * DD * 2);
    _Float16* qkv16 = (_Float16*)alloc((size_t)RR * 3 * DD * 2);
    _Float16* vT16  = (_Float16*)alloc((size_t)BH * HD * SS * 2);
    _Float16* p16   = (_Float16*)alloc((size_t)BH * SS * SS * 2);
    float*    att32 = (float*)alloc((size_t)BH * SS * SS * 4);
    _Float16* o16   = (_Float16*)alloc((size_t)RR * DD * 2);
    float*    x1    = (float*)alloc((size_t)RR * DD * 4);
    float*    ubuf  = (float*)alloc((size_t)RR * DFF * 4);
    _Float16* act16 = (_Float16*)alloc((size_t)RR * DFF * 2);
    float*    mu1   = (float*)alloc((size_t)RR * 4);
    float*    rs1   = (float*)alloc((size_t)RR * 4);
    float*    mu2   = (float*)alloc((size_t)RR * 4);
    float*    rs2   = (float*)alloc((size_t)RR * 4);
    _Float16* dqkv16= (_Float16*)alloc((size_t)RR * 3 * DD * 2);
    _Float16* dvT16 = (_Float16*)alloc((size_t)BH * HD * SS * 2);
    _Float16* dp16  = (_Float16*)alloc((size_t)BH * SS * SS * 2);
    float*    dxc   = (float*)alloc((size_t)RR * DD * 4);
    float*    dx1   = (float*)alloc((size_t)RR * DD * 4);
    float*    dof32 = (float*)alloc((size_t)RR * DD * 4);
    (void)ws_size; (void)in_sizes; (void)n_in; (void)out_size;

    auto gemmGrid = [](int M, int N, int Z) { return dim3((N + BN - 1) / BN, (M + BM - 1) / BM, Z); };
    auto mkP = [&]() { GemmParams p{}; p.zdiv = 1; p.scale = 1.0f; return p; };

    // ---- weight convert + transpose ----
    {
        long long t;
        t = (long long)DD * 3 * DD; cvtT_kernel<<<(t + 255) / 256, 256, 0, stream>>>(Wqkv, wqkvT, DD, 3 * DD, t);
        t = (long long)DD * DD;     cvtT_kernel<<<(t + 255) / 256, 256, 0, stream>>>(Wprj, wprjT, DD, DD, t);
        t = (long long)DD * DFF;    cvtT_kernel<<<(t + 255) / 256, 256, 0, stream>>>(W1, w1T, DD, DFF, t);
        t = (long long)DFF * DD;    cvtT_kernel<<<(t + 255) / 256, 256, 0, stream>>>(W2, w2T, DFF, DD, t);
    }

    // ================= PRIMAL =================
    ln_fwd_kernel<<<RR, 256, 0, stream>>>(x, g1, b1, h16, mu1, rs1, DD);

    { // qkv = h @ Wqkv   (f16 out)
        GemmParams p = mkP();
        p.A = h16; p.lda = DD; p.B = wqkvT; p.ldb = DD;
        p.Ch = qkv16; p.ldch = 3 * DD; p.M = RR; p.N = 3 * DD; p.K = DD;
        gemm_kernel<0, false, true><<<gemmGrid(RR, 3 * DD, 1), 256, 0, stream>>>(p);
    }
    { long long t = (long long)BH * HD * SS;
      vtrans_kernel<<<(t + 255) / 256, 256, 0, stream>>>(qkv16, vT16, 3 * DD, 2 * DD, t); }

    { // scores = scale * q @ k^T  (batched over b,h)
        GemmParams p = mkP();
        p.A = qkv16;        p.lda = 3 * DD; p.sA0 = (long long)SS * 3 * DD; p.sA1 = HD;
        p.B = qkv16 + DD;   p.ldb = 3 * DD; p.sB0 = (long long)SS * 3 * DD; p.sB1 = HD;
        p.Cf = att32; p.ldcf = SS; p.sCf0 = (long long)HH * SS * SS; p.sCf1 = (long long)SS * SS;
        p.M = SS; p.N = SS; p.K = HD; p.zdiv = HH; p.scale = SCALE_ATT;
        gemm_kernel<EPI_SCALE, true, false><<<gemmGrid(SS, SS, BH), 256, 0, stream>>>(p);
    }
    softmax_kernel<<<BH * SS, 128, 0, stream>>>(att32, p16, SS);

    { // o = p @ v  -> o16 interleaved [B,S,D]
        GemmParams p = mkP();
        p.A = p16;  p.lda = SS; p.sA0 = (long long)HH * SS * SS; p.sA1 = (long long)SS * SS;
        p.B = vT16; p.ldb = SS; p.sB0 = (long long)HH * HD * SS; p.sB1 = (long long)HD * SS;
        p.Ch = o16; p.ldch = DD; p.sCh0 = (long long)SS * DD; p.sCh1 = HD;
        p.M = SS; p.N = HD; p.K = SS; p.zdiv = HH;
        gemm_kernel<0, false, true><<<gemmGrid(SS, HD, BH), 256, 0, stream>>>(p);
    }
    { // x1 = x + o @ Wproj + bproj
        GemmParams p = mkP();
        p.A = o16; p.lda = DD; p.B = wprjT; p.ldb = DD;
        p.bias = bprj; p.resid = x; p.ldr = DD;
        p.Cf = x1; p.ldcf = DD; p.M = RR; p.N = DD; p.K = DD;
        gemm_kernel<EPI_BIAS | EPI_RESID, true, false><<<gemmGrid(RR, DD, 1), 256, 0, stream>>>(p);
    }
    ln_fwd_kernel<<<RR, 256, 0, stream>>>(x1, g2, b2, h16, mu2, rs2, DD);
    { // u = h2 @ W1 + bf1 ; act = gelu(u)
        GemmParams p = mkP();
        p.A = h16; p.lda = DD; p.B = w1T; p.ldb = DD; p.bias = bf1;
        p.Cf = ubuf; p.ldcf = DFF; p.Ch = act16; p.ldch = DFF;
        p.M = RR; p.N = DFF; p.K = DD;
        gemm_kernel<EPI_BIAS | EPI_GELU, true, true><<<gemmGrid(RR, DFF, 1), 256, 0, stream>>>(p);
    }
    { // out = x1 + act @ W2 + bf2
        GemmParams p = mkP();
        p.A = act16; p.lda = DFF; p.B = w2T; p.ldb = DFF; p.bias = bf2;
        p.resid = x1; p.ldr = DD; p.Cf = out; p.ldcf = DD;
        p.M = RR; p.N = DD; p.K = DFF;
        gemm_kernel<EPI_BIAS | EPI_RESID, true, false><<<gemmGrid(RR, DD, 1), 256, 0, stream>>>(p);
    }

    // ================= TANGENTS =================
    const long long tBSD = (long long)RR * DD;
    for (int m = 0; m < MT; ++m) {
        gather_kernel<<<(tBSD + 255) / 256, 256, 0, stream>>>(xt, dxc, m, tBSD);
        ln_jvp_kernel<<<RR, 256, 0, stream>>>(x, dxc, mu1, rs1, g1, h16, DD);
        { // dqkv = dh @ Wqkv
            GemmParams p = mkP();
            p.A = h16; p.lda = DD; p.B = wqkvT; p.ldb = DD;
            p.Ch = dqkv16; p.ldch = 3 * DD; p.M = RR; p.N = 3 * DD; p.K = DD;
            gemm_kernel<0, false, true><<<gemmGrid(RR, 3 * DD, 1), 256, 0, stream>>>(p);
        }
        { long long t = (long long)BH * HD * SS;
          vtrans_kernel<<<(t + 255) / 256, 256, 0, stream>>>(dqkv16, dvT16, 3 * DD, 2 * DD, t); }
        { // ds = scale * dq @ k^T
            GemmParams p = mkP();
            p.A = dqkv16;     p.lda = 3 * DD; p.sA0 = (long long)SS * 3 * DD; p.sA1 = HD;
            p.B = qkv16 + DD; p.ldb = 3 * DD; p.sB0 = (long long)SS * 3 * DD; p.sB1 = HD;
            p.Cf = att32; p.ldcf = SS; p.sCf0 = (long long)HH * SS * SS; p.sCf1 = (long long)SS * SS;
            p.M = SS; p.N = SS; p.K = HD; p.zdiv = HH; p.scale = SCALE_ATT;
            gemm_kernel<EPI_SCALE, true, false><<<gemmGrid(SS, SS, BH), 256, 0, stream>>>(p);
        }
        { // ds += scale * q @ dk^T
            GemmParams p = mkP();
            p.A = qkv16;       p.lda = 3 * DD; p.sA0 = (long long)SS * 3 * DD; p.sA1 = HD;
            p.B = dqkv16 + DD; p.ldb = 3 * DD; p.sB0 = (long long)SS * 3 * DD; p.sB1 = HD;
            p.Cf = att32; p.ldcf = SS; p.sCf0 = (long long)HH * SS * SS; p.sCf1 = (long long)SS * SS;
            p.M = SS; p.N = SS; p.K = HD; p.zdiv = HH; p.scale = SCALE_ATT;
            gemm_kernel<EPI_SCALE | EPI_ACC, true, false><<<gemmGrid(SS, SS, BH), 256, 0, stream>>>(p);
        }
        softmax_jvp_kernel<<<BH * SS, 128, 0, stream>>>(p16, att32, dp16, SS);
        { // do = dp @ v
            GemmParams p = mkP();
            p.A = dp16; p.lda = SS; p.sA0 = (long long)HH * SS * SS; p.sA1 = (long long)SS * SS;
            p.B = vT16; p.ldb = SS; p.sB0 = (long long)HH * HD * SS; p.sB1 = (long long)HD * SS;
            p.Cf = dof32; p.ldcf = DD; p.sCf0 = (long long)SS * DD; p.sCf1 = HD;
            p.M = SS; p.N = HD; p.K = SS; p.zdiv = HH;
            gemm_kernel<0, true, false><<<gemmGrid(SS, HD, BH), 256, 0, stream>>>(p);
        }
        { // do += p @ dv  -> f16
            GemmParams p = mkP();
            p.A = p16;   p.lda = SS; p.sA0 = (long long)HH * SS * SS; p.sA1 = (long long)SS * SS;
            p.B = dvT16; p.ldb = SS; p.sB0 = (long long)HH * HD * SS; p.sB1 = (long long)HD * SS;
            p.Cf = dof32; p.ldcf = DD; p.sCf0 = (long long)SS * DD; p.sCf1 = HD;
            p.Ch = o16;   p.ldch = DD; p.sCh0 = (long long)SS * DD; p.sCh1 = HD;
            p.M = SS; p.N = HD; p.K = SS; p.zdiv = HH;
            gemm_kernel<EPI_ACC, false, true><<<gemmGrid(SS, HD, BH), 256, 0, stream>>>(p);
        }
        { // dx1 = dx + do @ Wproj
            GemmParams p = mkP();
            p.A = o16; p.lda = DD; p.B = wprjT; p.ldb = DD;
            p.resid = dxc; p.ldr = DD; p.Cf = dx1; p.ldcf = DD;
            p.M = RR; p.N = DD; p.K = DD;
            gemm_kernel<EPI_RESID, true, false><<<gemmGrid(RR, DD, 1), 256, 0, stream>>>(p);
        }
        ln_jvp_kernel<<<RR, 256, 0, stream>>>(x1, dx1, mu2, rs2, g2, h16, DD);
        { // dact = gelu'(u) * (dh2 @ W1)
            GemmParams p = mkP();
            p.A = h16; p.lda = DD; p.B = w1T; p.ldb = DD;
            p.U = ubuf; p.ldu = DFF; p.Ch = act16; p.ldch = DFF;
            p.M = RR; p.N = DFF; p.K = DD;
            gemm_kernel<EPI_GPRIME, false, true><<<gemmGrid(RR, DFF, 1), 256, 0, stream>>>(p);
        }
        { // dout = dx1 + dact @ W2   (in-place into dx1)
            GemmParams p = mkP();
            p.A = act16; p.lda = DFF; p.B = w2T; p.ldb = DFF;
            p.resid = dx1; p.ldr = DD; p.Cf = dx1; p.ldcf = DD;
            p.M = RR; p.N = DD; p.K = DFF;
            gemm_kernel<EPI_RESID, true, false><<<gemmGrid(RR, DD, 1), 256, 0, stream>>>(p);
        }
        scatter_kernel<<<(tBSD + 255) / 256, 256, 0, stream>>>(dx1, out + tBSD, m, tBSD);
    }
}
